// Graph_GRU_7327214207533
// MI455X (gfx1250) — matrix operand order repaired and checked by reading, not compiled, not run on hardware
//
#include <hip/hip_runtime.h>
#include <hip/hip_bf16.h>
#include <math.h>

#define NN 50000
#define EE 1600000
#define DD 128
#define LL 2
#define MTILES (NN / 16)   // 3125 row tiles (exact)

typedef __attribute__((ext_vector_type(16))) __bf16 v16bf;
typedef __attribute__((ext_vector_type(8)))  float  v8f;

// ---------------------------------------------------------------------------
// Degree / normalization
// ---------------------------------------------------------------------------
__global__ __launch_bounds__(256) void deg_kernel(const int* __restrict__ dst,
                                                  float* __restrict__ deg) {
    int e = blockIdx.x * 256 + threadIdx.x;
    if (e < EE) atomicAdd(&deg[dst[e]], 1.0f);
}

__global__ __launch_bounds__(256) void dinv_kernel(const float* __restrict__ deg,
                                                   float* __restrict__ dinv) {
    int i = blockIdx.x * 256 + threadIdx.x;
    if (i < NN) dinv[i] = rsqrtf(deg[i] + 1.0f);
}

// agg[i,:] = X[i,:] * dinv[i]^2   (self-loop term; also initializes the buffer)
__global__ __launch_bounds__(256) void selfinit(const float* __restrict__ X,
                                                const float* __restrict__ dinv,
                                                float* __restrict__ agg) {
    int i = blockIdx.x * 256 + threadIdx.x;
    if (i >= NN * DD) return;
    float di = dinv[i >> 7];
    agg[i] = X[i] * di * di;
}

// ---------------------------------------------------------------------------
// Edge scatter: agg[dst] += X[src] * dinv[src]*dinv[dst]
// One wave (32 lanes) per edge, 4 floats per lane.
// ---------------------------------------------------------------------------
__global__ __launch_bounds__(256) void scatter2(const int* __restrict__ src,
                                                const int* __restrict__ dst,
                                                const float* __restrict__ dinv,
                                                const float* __restrict__ X,
                                                const float* __restrict__ H,
                                                float* __restrict__ aggX,
                                                float* __restrict__ aggH) {
    int e = blockIdx.x * 8 + (threadIdx.x >> 5);
    if (e >= EE) return;
    int lane = threadIdx.x & 31;
    int s = src[e], d = dst[e];
    float c = dinv[s] * dinv[d];
    int k = lane * 4;
    const float4 xv = *(const float4*)(X + (size_t)s * DD + k);
    const float4 hv = *(const float4*)(H + (size_t)s * DD + k);
    float* px = aggX + (size_t)d * DD + k;
    float* ph = aggH + (size_t)d * DD + k;
    atomicAdd(px + 0, xv.x * c); atomicAdd(px + 1, xv.y * c);
    atomicAdd(px + 2, xv.z * c); atomicAdd(px + 3, xv.w * c);
    atomicAdd(ph + 0, hv.x * c); atomicAdd(ph + 1, hv.y * c);
    atomicAdd(ph + 2, hv.z * c); atomicAdd(ph + 3, hv.w * c);
}

__global__ __launch_bounds__(256) void scatter1(const int* __restrict__ src,
                                                const int* __restrict__ dst,
                                                const float* __restrict__ dinv,
                                                const float* __restrict__ X,
                                                float* __restrict__ aggX) {
    int e = blockIdx.x * 8 + (threadIdx.x >> 5);
    if (e >= EE) return;
    int lane = threadIdx.x & 31;
    int s = src[e], d = dst[e];
    float c = dinv[s] * dinv[d];
    int k = lane * 4;
    const float4 xv = *(const float4*)(X + (size_t)s * DD + k);
    float* px = aggX + (size_t)d * DD + k;
    atomicAdd(px + 0, xv.x * c); atomicAdd(px + 1, xv.y * c);
    atomicAdd(px + 2, xv.z * c); atomicAdd(px + 3, xv.w * c);
}

// ---------------------------------------------------------------------------
// Weight pre-pack: f32 [128][128] row-major -> bf16 WMMA B-operand register
// image. Tile index = (((l*2+which)*3+gate)*4+kb)*8+t, 512 bf16 per tile.
// Per lane (n = t*16 + lane&15, khalf = lane>>4): e[0..7] = K 8*khalf+0..7, e[8..15] = K 16+8*khalf+0..7.
// ---------------------------------------------------------------------------
__global__ __launch_bounds__(32) void prepack_w(const float* __restrict__ Wx,
                                                const float* __restrict__ Wh,
                                                __bf16* __restrict__ Wp) {
    int tile = blockIdx.x;
    int lane = threadIdx.x;
    int t = tile & 7;
    int r = tile >> 3;
    int kb = r & 3;   r >>= 2;
    int gate = r % 3; r /= 3;
    int which = r & 1;
    int l = r >> 1;
    const float* W = (which ? Wh : Wx) + (size_t)((l * 3 + gate) * DD) * DD;
    int n  = t * 16 + (lane & 15);
    int k0 = kb * 32 + (lane >> 4) * 8;
    __bf16* dstp = Wp + (size_t)tile * 512 + lane * 16;
#pragma unroll
    for (int i = 0; i < 16; ++i)
        dstp[i] = (__bf16)W[(size_t)(k0 + (i < 8 ? i : i + 8)) * DD + n];
}

// ---------------------------------------------------------------------------
// A-operand load: 16x32 bf16 tile from a row-major f32 matrix.
// Lane layout (ISA 7.12.2): e[0..7] = K = kb*32 + 8*half + 0..7,
//                           e[8..15] = K = kb*32 + 16 + 8*half + 0..7.
// ---------------------------------------------------------------------------
__device__ __forceinline__ v16bf load_a_tile(const float* __restrict__ rowp,
                                             int kb, int half) {
    const float* p0 = rowp + kb * 32 + half * 8;
    const float* p1 = p0 + 16;
    float4 q0 = *(const float4*)(p0);
    float4 q1 = *(const float4*)(p0 + 4);
    float4 q2 = *(const float4*)(p1);
    float4 q3 = *(const float4*)(p1 + 4);
    v16bf a;
    a[0]  = (__bf16)q0.x; a[1]  = (__bf16)q0.y; a[2]  = (__bf16)q0.z; a[3]  = (__bf16)q0.w;
    a[4]  = (__bf16)q1.x; a[5]  = (__bf16)q1.y; a[6]  = (__bf16)q1.z; a[7]  = (__bf16)q1.w;
    a[8]  = (__bf16)q2.x; a[9]  = (__bf16)q2.y; a[10] = (__bf16)q2.z; a[11] = (__bf16)q2.w;
    a[12] = (__bf16)q3.x; a[13] = (__bf16)q3.y; a[14] = (__bf16)q3.z; a[15] = (__bf16)q3.w;
    return a;
}

__device__ __forceinline__ float sigmoidf_dev(float x) {
    return 1.0f / (1.0f + __expf(-x));
}

// ---------------------------------------------------------------------------
// Gates z and r:  g = sigmoid(aggX @ Wx[g] + bx[g] + aggH @ Wh[g] + bh[g])
// z -> z_out ; r -> rh_out = r * h   (rh_out may alias aggH: each wave fully
// reads its 16 rows into registers before writing them).
// One wave per 16-row strip, 8 waves per block; 128 WMMAs per wave.
// ---------------------------------------------------------------------------
__global__ __launch_bounds__(256)
void gru_gates_zr(const float* __restrict__ aggX, const float* __restrict__ aggH,
                  const __bf16* __restrict__ Wp,
                  const float* __restrict__ bx_l, const float* __restrict__ bh_l,
                  const float* __restrict__ hl,
                  float* __restrict__ z_out, float* __restrict__ rh_out,
                  int layer) {
    int wave = threadIdx.x >> 5;
    int lane = threadIdx.x & 31;
    int tile_m = blockIdx.x * 8 + wave;
    if (tile_m >= MTILES) return;
    int m0   = tile_m * 16;
    int half = lane >> 4;
    int row  = m0 + (lane & 15);

    const float* rx = aggX + (size_t)row * DD;
    const float* rh = aggH + (size_t)row * DD;
    v16bf ax[4], ah[4];
#pragma unroll
    for (int kb = 0; kb < 4; ++kb) {
        ax[kb] = load_a_tile(rx, kb, half);
        ah[kb] = load_a_tile(rh, kb, half);
    }

#pragma unroll
    for (int gate = 0; gate < 2; ++gate) {
        int txb = (((layer * 2 + 0) * 3 + gate) * 4) * 8;  // x-stream tile base
        int thb = (((layer * 2 + 1) * 3 + gate) * 4) * 8;  // h-stream tile base
#pragma unroll
        for (int t = 0; t < 8; ++t) {
            v8f acc = {};
#pragma unroll
            for (int kb = 0; kb < 4; ++kb) {
                v16bf bxt = *(const v16bf*)(Wp + (size_t)(txb + kb * 8 + t) * 512 + lane * 16);
                acc = __builtin_amdgcn_wmma_f32_16x16x32_bf16(false, ax[kb], false, bxt,
                                                              (short)0, acc, false, false);
                v16bf bht = *(const v16bf*)(Wp + (size_t)(thb + kb * 8 + t) * 512 + lane * 16);
                acc = __builtin_amdgcn_wmma_f32_16x16x32_bf16(false, ah[kb], false, bht,
                                                              (short)0, acc, false, false);
            }
            int col = t * 16 + (lane & 15);
            float bias = bx_l[gate * DD + col] + bh_l[gate * DD + col];
#pragma unroll
            for (int v = 0; v < 8; ++v) {
                int r = m0 + half * 8 + v;   // C layout: M = v + 8*half
                float val = sigmoidf_dev(acc[v] + bias);
                if (gate == 0) z_out[(size_t)r * DD + col] = val;
                else           rh_out[(size_t)r * DD + col] = val * hl[(size_t)r * DD + col];
            }
        }
    }
}

// ---------------------------------------------------------------------------
// Candidate + combine:
//   ht  = tanh(aggX @ Wx[2] + bx[2] + aggRH @ Wh[2] + bh[2])
//   out = z*h + (1-z)*ht
// ---------------------------------------------------------------------------
__global__ __launch_bounds__(256)
void gru_final(const float* __restrict__ aggX, const float* __restrict__ aggRH,
               const __bf16* __restrict__ Wp,
               const float* __restrict__ bx_l, const float* __restrict__ bh_l,
               const float* __restrict__ hl, const float* __restrict__ zb,
               float* __restrict__ outl, int layer) {
    int wave = threadIdx.x >> 5;
    int lane = threadIdx.x & 31;
    int tile_m = blockIdx.x * 8 + wave;
    if (tile_m >= MTILES) return;
    int m0   = tile_m * 16;
    int half = lane >> 4;
    int row  = m0 + (lane & 15);

    const float* rx = aggX  + (size_t)row * DD;
    const float* rc = aggRH + (size_t)row * DD;
    v16bf ax[4], ac[4];
#pragma unroll
    for (int kb = 0; kb < 4; ++kb) {
        ax[kb] = load_a_tile(rx, kb, half);
        ac[kb] = load_a_tile(rc, kb, half);
    }

    const int gate = 2;
    int txb = (((layer * 2 + 0) * 3 + gate) * 4) * 8;
    int thb = (((layer * 2 + 1) * 3 + gate) * 4) * 8;
#pragma unroll
    for (int t = 0; t < 8; ++t) {
        v8f acc = {};
#pragma unroll
        for (int kb = 0; kb < 4; ++kb) {
            v16bf bxt = *(const v16bf*)(Wp + (size_t)(txb + kb * 8 + t) * 512 + lane * 16);
            acc = __builtin_amdgcn_wmma_f32_16x16x32_bf16(false, ax[kb], false, bxt,
                                                          (short)0, acc, false, false);
            v16bf bht = *(const v16bf*)(Wp + (size_t)(thb + kb * 8 + t) * 512 + lane * 16);
            acc = __builtin_amdgcn_wmma_f32_16x16x32_bf16(false, ac[kb], false, bht,
                                                          (short)0, acc, false, false);
        }
        int col = t * 16 + (lane & 15);
        float bias = bx_l[gate * DD + col] + bh_l[gate * DD + col];
#pragma unroll
        for (int v = 0; v < 8; ++v) {
            int r = m0 + half * 8 + v;
            float ht = tanhf(acc[v] + bias);
            float zz = zb[(size_t)r * DD + col];
            float hv = hl[(size_t)r * DD + col];
            outl[(size_t)r * DD + col] = zz * hv + (1.0f - zz) * ht;
        }
    }
}

// ---------------------------------------------------------------------------
// Host launch
// ---------------------------------------------------------------------------
extern "C" void kernel_launch(void* const* d_in, const int* in_sizes, int n_in,
                              void* d_out, int out_size, void* d_ws, size_t ws_size,
                              hipStream_t stream) {
    (void)in_sizes; (void)n_in; (void)out_size; (void)ws_size;

    const float* x  = (const float*)d_in[0];
    const int*   ei = (const int*)d_in[1];
    const float* h  = (const float*)d_in[2];
    const float* Wx = (const float*)d_in[3];
    const float* bx = (const float*)d_in[4];
    const float* Wh = (const float*)d_in[5];
    const float* bh = (const float*)d_in[6];
    float* out = (float*)d_out;

    const int* src = ei;
    const int* dst = ei + EE;

    char* ws = (char*)d_ws;
    float* deg   = (float*)ws;  ws += (size_t)NN * sizeof(float);
    float* dinv  = (float*)ws;  ws += (size_t)NN * sizeof(float);
    float* aggX  = (float*)ws;  ws += (size_t)NN * DD * sizeof(float);
    float* aggH  = (float*)ws;  ws += (size_t)NN * DD * sizeof(float);  // reused as r*h
    float* aggRH = (float*)ws;  ws += (size_t)NN * DD * sizeof(float);
    float* zbuf  = (float*)ws;  ws += (size_t)NN * DD * sizeof(float);
    __bf16* Wp   = (__bf16*)ws;                       // 384 tiles * 512 bf16 = 384 KB

    const int thr = 256;
    hipMemsetAsync(deg, 0, (size_t)NN * sizeof(float), stream);
    deg_kernel <<<(EE + thr - 1) / thr, thr, 0, stream>>>(dst, deg);
    dinv_kernel<<<(NN + thr - 1) / thr, thr, 0, stream>>>(deg, dinv);
    prepack_w  <<<384, 32, 0, stream>>>(Wx, Wh, Wp);

    const int gemm_grid = (MTILES + 7) / 8;
    const int elem_grid = (NN * DD + thr - 1) / thr;
    const int edge_grid = (EE + 7) / 8;

    for (int l = 0; l < LL; ++l) {
        const float* inp = (l == 0) ? x : out;             // layer-0 output feeds layer 1
        const float* hl  = h + (size_t)l * NN * DD;
        float* outl      = out + (size_t)l * NN * DD;

        selfinit<<<elem_grid, thr, 0, stream>>>(inp, dinv, aggX);
        selfinit<<<elem_grid, thr, 0, stream>>>(hl,  dinv, aggH);
        scatter2<<<edge_grid, thr, 0, stream>>>(src, dst, dinv, inp, hl, aggX, aggH);

        // z -> zbuf ; r*h -> aggH (safe in-place: rows read to regs first)
        gru_gates_zr<<<gemm_grid, thr, 0, stream>>>(aggX, aggH, Wp,
                                                    bx + (size_t)l * 3 * DD,
                                                    bh + (size_t)l * 3 * DD,
                                                    hl, zbuf, aggH, l);

        selfinit<<<elem_grid, thr, 0, stream>>>(aggH, dinv, aggRH);
        scatter1<<<edge_grid, thr, 0, stream>>>(src, dst, dinv, aggH, aggRH);

        gru_final<<<gemm_grid, thr, 0, stream>>>(aggX, aggRH, Wp,
                                                 bx + (size_t)l * 3 * DD,
                                                 bh + (size_t)l * 3 * DD,
                                                 hl, zbuf, outl, l);
    }
}
